// BitFFN_31224412242443
// MI455X (gfx1250) — 8-bit matrix operand order repaired and checked by reading, compiled, not run on hardware
//
#include <hip/hip_runtime.h>
#include <hip/hip_bf16.h>
#include <cstdint>

typedef __attribute__((ext_vector_type(8))) int v8i;

typedef int v4i_vec __attribute__((vector_size(16)));
typedef __attribute__((address_space(1))) v4i_vec as1_v4i;
typedef __attribute__((address_space(3))) v4i_vec as3_v4i;

#define EPSF 1e-5f
#define QBF  127.0f

#define BM 128
#define BN 128
#define BK 64
#define LDSS 80
#define STAGE ((BM + BN) * LDSS)

#if __has_builtin(__builtin_amdgcn_global_load_async_to_lds_b128)
#define HAVE_ASYNC_LDS 1
#else
#define HAVE_ASYNC_LDS 0
#endif

#if __has_builtin(__builtin_amdgcn_s_wait_asynccnt)
#define WAIT_ASYNC(n) __builtin_amdgcn_s_wait_asynccnt(n)
#else
#define WAIT_ASYNC(n) asm volatile("s_wait_asynccnt %0" ::"i"(n) : "memory")
#endif

__global__ __launch_bounds__(256) void k_ln_stats(const float* __restrict__ x,
                                                  const float* __restrict__ lw,
                                                  const float* __restrict__ lb,
                                                  float* __restrict__ stats,
                                                  unsigned* __restrict__ amax) {
  __shared__ float red[256];
  const int tid = threadIdx.x;
  const size_t tok = blockIdx.x;
  const float* xr = x + tok * 2048;

  float vals[8];
  float s = 0.f;
#pragma unroll
  for (int i = 0; i < 8; ++i) { vals[i] = xr[tid + i * 256]; s += vals[i]; }
  red[tid] = s; __syncthreads();
  for (int st = 128; st > 0; st >>= 1) { if (tid < st) red[tid] += red[tid + st]; __syncthreads(); }
  const float mu = red[0] * (1.0f / 2048.0f);
  __syncthreads();

  float v = 0.f;
#pragma unroll
  for (int i = 0; i < 8; ++i) { float d = vals[i] - mu; v += d * d; }
  red[tid] = v; __syncthreads();
  for (int st = 128; st > 0; st >>= 1) { if (tid < st) red[tid] += red[tid + st]; __syncthreads(); }
  const float rs = rsqrtf(red[0] * (1.0f / 2048.0f) + EPSF);
  __syncthreads();

  float m = 0.f;
#pragma unroll
  for (int i = 0; i < 8; ++i) {
    int d = tid + i * 256;
    float ln = (vals[i] - mu) * rs * lw[d] + lb[d];
    m = fmaxf(m, fabsf(ln));
  }
  red[tid] = m; __syncthreads();
  for (int st = 128; st > 0; st >>= 1) { if (tid < st) red[tid] = fmaxf(red[tid], red[tid + st]); __syncthreads(); }
  if (tid == 0) {
    stats[2 * tok]     = mu;
    stats[2 * tok + 1] = rs;
    atomicMax(amax, __float_as_uint(red[0]));
  }
}

__global__ __launch_bounds__(256) void k_abssum(const float* __restrict__ w, long n,
                                                float* __restrict__ out) {
  __shared__ float red[256];
  const int tid = threadIdx.x;
  float s = 0.f;
  for (long i = (long)blockIdx.x * 256 + tid; i < n; i += (long)gridDim.x * 256)
    s += fabsf(w[i]);
  red[tid] = s; __syncthreads();
  for (int st = 128; st > 0; st >>= 1) { if (tid < st) red[tid] += red[tid + st]; __syncthreads(); }
  if (tid == 0) atomicAdd(out, red[0]);
}

__global__ __launch_bounds__(256) void k_quant_w(const float* __restrict__ w,
                                                 const float* __restrict__ sumptr,
                                                 float invCount,
                                                 int8_t* __restrict__ q) {
  const float beta = fmaxf(*sumptr * invCount, EPSF);
  const float ib = 1.0f / beta;
  const long i = (long)blockIdx.x * 256 + threadIdx.x;
  float v = rintf(w[i] * ib);
  v = fminf(fmaxf(v, -1.0f), 1.0f);
  q[i] = (int8_t)(int)v;
}

__global__ __launch_bounds__(256) void k_quant_x(const float* __restrict__ x,
                                                 const float* __restrict__ lw,
                                                 const float* __restrict__ lb,
                                                 const float* __restrict__ stats,
                                                 const float* __restrict__ scal,
                                                 int8_t* __restrict__ q) {
  const float maxa = __uint_as_float(((const unsigned*)scal)[0]);
  const float scale = QBF / fmaxf(maxa, EPSF);
  const long i = (long)blockIdx.x * 256 + threadIdx.x;
  const long t = i >> 11;
  const int  d = (int)(i & 2047);
  const float v = (x[i] - stats[2 * t]) * stats[2 * t + 1] * lw[d] + lb[d];
  float qv = fminf(fmaxf(rintf(v * scale), -QBF), QBF);
  q[i] = (int8_t)(int)qv;
}

__global__ __launch_bounds__(256) void k_quant_h(const __hip_bfloat16* __restrict__ h,
                                                 const float* __restrict__ scal,
                                                 int8_t* __restrict__ q) {
  const float maxh = __uint_as_float(((const unsigned*)scal)[3]);
  const float scale = QBF / fmaxf(maxh, EPSF);
  const long i = (long)blockIdx.x * 256 + threadIdx.x;
  const float v = __bfloat162float(h[i]);
  float qv = fminf(fmaxf(rintf(v * scale), -QBF), QBF);
  q[i] = (int8_t)(int)qv;
}

template <int MODE>
__global__ __launch_bounds__(256) void gemm_iu8(const int8_t* __restrict__ A,
                                                const int8_t* __restrict__ Bm,
                                                void* __restrict__ Out,
                                                int M, int N, int K,
                                                float* __restrict__ scal,
                                                float invWCount) {
  __shared__ int4 lds4[2 * STAGE / 16];
  char* lds = (char*)lds4;

  const int tid  = threadIdx.x;
  const int lane = tid & 31;
  const int wave = tid >> 5;
  const int wm   = wave >> 2;
  const int wn   = wave & 3;

  const int bm = blockIdx.y * BM;
  const int bn = blockIdx.x * BN;

  v8i acc[4][2];
#pragma unroll
  for (int i = 0; i < 4; ++i)
#pragma unroll
    for (int j = 0; j < 2; ++j)
#pragma unroll
      for (int g = 0; g < 8; ++g) acc[i][j][g] = 0;

  const int  lrow  = tid >> 1;
  const int  lhalf = (tid & 1) * 32;
  const int8_t* gA = A  + (long)(bm + lrow) * K + lhalf;
  const int8_t* gB = Bm + (long)(bn + lrow) * K + lhalf;
  char* stA = lds + lrow * LDSS + lhalf;
  char* stB = lds + BM * LDSS + lrow * LDSS + lhalf;

  const int afr = lane & 15;
  const int akb = (lane >> 4) * 16;
  const int bkb = (lane >> 4) * 16;

  const int nk = K / BK;

#if HAVE_ASYNC_LDS
  auto issue = [&](int kc, int buf) {
    __builtin_amdgcn_global_load_async_to_lds_b128(
        (as1_v4i*)(gA + kc),      (as3_v4i*)(stA + buf * STAGE),      0, 0);
    __builtin_amdgcn_global_load_async_to_lds_b128(
        (as1_v4i*)(gA + kc + 16), (as3_v4i*)(stA + buf * STAGE + 16), 0, 0);
    __builtin_amdgcn_global_load_async_to_lds_b128(
        (as1_v4i*)(gB + kc),      (as3_v4i*)(stB + buf * STAGE),      0, 0);
    __builtin_amdgcn_global_load_async_to_lds_b128(
        (as1_v4i*)(gB + kc + 16), (as3_v4i*)(stB + buf * STAGE + 16), 0, 0);
  };
  issue(0, 0);
#endif

  for (int ic = 0; ic < nk; ++ic) {
    const int cur = ic & 1;
    char* ldsA = lds + cur * STAGE * HAVE_ASYNC_LDS;
    char* ldsB = ldsA + BM * LDSS;

#if HAVE_ASYNC_LDS
    if (ic + 1 < nk) {
      issue((ic + 1) * BK, cur ^ 1);
      WAIT_ASYNC(4);
    } else {
      WAIT_ASYNC(0);
    }
    __syncthreads();
#else
    __syncthreads();
    const int kc = ic * BK;
    const int4* ga = (const int4*)(gA + kc);
    const int4* gb = (const int4*)(gB + kc);
    *(int4*)(stA)      = ga[0];
    *(int4*)(stA + 16) = ga[1];
    *(int4*)(stB)      = gb[0];
    *(int4*)(stB + 16) = gb[1];
    __syncthreads();
#endif

    v8i bfrag[2];
#pragma unroll
    for (int j = 0; j < 2; ++j) {
      const char* pb = ldsB + (wn * 32 + j * 16 + afr) * LDSS + bkb;
      const int* q0 = (const int*)pb;
      const int* q1 = (const int*)(pb + 32);
#pragma unroll
      for (int g = 0; g < 4; ++g) { bfrag[j][g] = q0[g]; bfrag[j][4 + g] = q1[g]; }
    }
#pragma unroll
    for (int i = 0; i < 4; ++i) {
      const char* pa = ldsA + (wm * 64 + i * 16 + afr) * LDSS + akb;
      v8i af;
#pragma unroll
      for (int g = 0; g < 4; ++g) {
        const int* p2 = (const int*)(pa + (g >> 1) * 32 + (g & 1) * 8);
        af[2 * g]     = p2[0];
        af[2 * g + 1] = p2[1];
      }
#pragma unroll
      for (int j = 0; j < 2; ++j) {
        acc[i][j] = __builtin_amdgcn_wmma_i32_16x16x64_iu8(
            true, af, true, bfrag[j], acc[i][j], false, false);
      }
    }
    __syncthreads();
  }

  const unsigned* su = (const unsigned*)scal;
  const int hi8 = (lane >> 4) << 3;
  const int nn  = lane & 15;

  if (MODE == 1) {
    const float maxa = __uint_as_float(su[0]);
    const float beta = fmaxf(scal[1] * invWCount, EPSF);
    const float dq   = fmaxf(maxa, EPSF) * (1.0f / QBF) * beta;
    __hip_bfloat16* H = (__hip_bfloat16*)Out;
    float lmax = 0.f;
#pragma unroll
    for (int i = 0; i < 4; ++i)
#pragma unroll
      for (int j = 0; j < 2; ++j) {
        const int m0 = bm + wm * 64 + i * 16 + hi8;
        const int n  = bn + wn * 32 + j * 16 + nn;
#pragma unroll
        for (int g = 0; g < 8; ++g) {
          float v  = (float)acc[i][j][g] * dq;
          float gl = 0.5f * v * (1.0f + erff(v * 0.70710678118654752f));
          H[(size_t)(m0 + g) * N + n] = __float2bfloat16(gl);
          lmax = fmaxf(lmax, fabsf(gl));
        }
      }
    __syncthreads();
    float* red = (float*)lds4;
    red[tid] = lmax; __syncthreads();
    for (int st = 128; st > 0; st >>= 1) { if (tid < st) red[tid] = fmaxf(red[tid], red[tid + st]); __syncthreads(); }
    if (tid == 0) atomicMax((unsigned*)(scal + 3), __float_as_uint(red[0]));
  } else {
    const float maxh  = __uint_as_float(su[3]);
    const float beta2 = fmaxf(scal[2] * invWCount, EPSF);
    const float dq    = fmaxf(maxh, EPSF) * (1.0f / QBF) * beta2;
    float* O = (float*)Out;
#pragma unroll
    for (int i = 0; i < 4; ++i)
#pragma unroll
      for (int j = 0; j < 2; ++j) {
        const int m0 = bm + wm * 64 + i * 16 + hi8;
        const int n  = bn + wn * 32 + j * 16 + nn;
#pragma unroll
        for (int g = 0; g < 8; ++g)
          O[(size_t)(m0 + g) * N + n] = (float)acc[i][j][g] * dq;
      }
  }
}

extern "C" void kernel_launch(void* const* d_in, const int* in_sizes, int n_in,
                              void* d_out, int out_size, void* d_ws, size_t ws_size,
                              hipStream_t stream) {
  (void)in_sizes; (void)n_in; (void)out_size; (void)ws_size;
  const float* x   = (const float*)d_in[0];
  const float* lnw = (const float*)d_in[1];
  const float* lnb = (const float*)d_in[2];
  const float* w1  = (const float*)d_in[3];
  const float* w2  = (const float*)d_in[4];
  float* out = (float*)d_out;

  const int M = 8192;
  const int D = 2048;
  const int F = 8192;

  char* p = (char*)d_ws;
  float*  scal  = (float*)p;
  float*  stats = (float*)(p + 256);
  int8_t* xq    = (int8_t*)(p + 256 + (size_t)2 * M * 4);
  int8_t* w1q   = xq  + (size_t)M * D;
  int8_t* w2q   = w1q + (size_t)F * D;
  __hip_bfloat16* h = (__hip_bfloat16*)(w2q + (size_t)D * F);
  int8_t* hq    = (int8_t*)((char*)h + (size_t)M * F * 2);

  (void)hipMemsetAsync(d_ws, 0, 256, stream);

  k_ln_stats<<<M, 256, 0, stream>>>(x, lnw, lnb, stats, (unsigned*)scal);
  k_abssum<<<2048, 256, 0, stream>>>(w1, (long)F * D, scal + 1);
  k_abssum<<<2048, 256, 0, stream>>>(w2, (long)D * F, scal + 2);

  const float invW = 1.0f / (float)((long)F * D);
  k_quant_x<<<(M * D) / 256, 256, 0, stream>>>(x, lnw, lnb, stats, scal, xq);
  k_quant_w<<<(F * D) / 256, 256, 0, stream>>>(w1, scal + 1, invW, w1q);
  k_quant_w<<<(D * F) / 256, 256, 0, stream>>>(w2, scal + 2, invW, w2q);

  dim3 g1(F / BN, M / BM);
  gemm_iu8<1><<<g1, 256, 0, stream>>>(xq, w1q, (void*)h, M, F, D, scal, invW);

  k_quant_h<<<((long)M * F) / 256, 256, 0, stream>>>(h, scal, hq);

  dim3 g2(D / BN, M / BM);
  gemm_iu8<2><<<g2, 256, 0, stream>>>(hq, w2q, (void*)out, M, D, F, scal, invW);
}
